// LineVectorizer_10342281249173
// MI455X (gfx1250) — matrix operand order repaired and checked by reading, not compiled, not run on hardware
//
#include <hip/hip_runtime.h>
#include <hip/hip_bf16.h>
#include <cstdint>
#include <cstddef>

// ---------------------------------------------------------------------------
// LOI head for gfx1250 (MI455X): bf16 WMMA GEMMs, async-LDS double buffering,
// L2-resident bilinear gather, coalesced b128 epilogue stores.
// ---------------------------------------------------------------------------

typedef __attribute__((ext_vector_type(16))) __bf16 v16bf;
typedef __attribute__((ext_vector_type(8)))  __bf16 v8bf;
typedef __attribute__((ext_vector_type(8)))  float  v8f;
typedef int v4i_gcc __attribute__((vector_size(16)));   // matches builtin param

static __device__ __forceinline__ __bf16 f2b(float f) { return (__bf16)f; }
static __device__ __forceinline__ float  b2f(__bf16 b) { return (float)b; }

#define HW      128
#define DIM_LOI 128
#define DIM_FC  1024
#define N_PTS0  32
#define N_PTS1  8
#define KZ      1056   // 1032 padded up to a multiple of 32
#define LDSA    40     // padded LDS row stride (bf16 elems) for A tile (128x32)
#define LDSB    40     // padded LDS row stride for Bt tile (64x32)
#define BUFE    (128 * LDSA + 64 * LDSB)   // elems per double-buffer half

// ---- CDNA5 async global->LDS path (probe via __has_builtin; safe fallback) --
#if defined(__has_builtin)
#  if __has_builtin(__builtin_amdgcn_global_load_async_to_lds_b128)
#    define HAVE_ASYNC_LDS 1
#  endif
#  if __has_builtin(__builtin_amdgcn_s_wait_asynccnt)
#    define HAVE_WAIT_ASYNC 1
#  endif
#endif

#if defined(HAVE_ASYNC_LDS)
static __device__ __forceinline__ void async_copy16(const void* g, void* l)
{
    // Builtin takes generic int4* params (probe-confirmed); the compiler
    // resolves the flat->global / flat->LDS address spaces itself.
    __builtin_amdgcn_global_load_async_to_lds_b128(
        (v4i_gcc*)g, (v4i_gcc*)l, 0, 0);
}
static __device__ __forceinline__ void wait_async0()
{
#if defined(HAVE_WAIT_ASYNC)
    __builtin_amdgcn_s_wait_asynccnt(0);
#else
    asm volatile("s_wait_asynccnt 0x0" ::: "memory");
#endif
}
#endif

// -------------------------- weight prep (f32 -> bf16, transpose) -----------
__global__ void prep_weights_kernel(const float* __restrict__ Wfc1,
                                    const float* __restrict__ W1,
                                    const float* __restrict__ W2,
                                    __bf16* __restrict__ wfc1b,   // [128][256]
                                    __bf16* __restrict__ w1t,     // [1024][1056] n-major
                                    __bf16* __restrict__ w2t)     // [1024][1024] n-major
{
    const int T1 = DIM_LOI * 256;
    const int T2 = DIM_FC * KZ;
    int idx = blockIdx.x * blockDim.x + threadIdx.x;
    const int total = T1 + T2 + DIM_FC * DIM_FC;
    if (idx >= total) return;
    if (idx < T1) { wfc1b[idx] = f2b(Wfc1[idx]); return; }
    idx -= T1;
    if (idx < T2) {
        int n = idx / KZ, k = idx % KZ;
        w1t[idx] = (k < 1032) ? f2b(W1[(size_t)k * DIM_FC + n]) : (__bf16)0.0f;
        return;
    }
    idx -= T2;
    int n = idx >> 10, k = idx & 1023;
    w2t[idx] = f2b(W2[((size_t)k << 10) + n]);
}

// -------------------------- WMMA fragment loaders (gfx1250 layouts) --------
// A 16x32 bf16: lane<16 -> row=lane; elems 0..7 = K[kh*8..+7],
// elems 8..15 = K[kh*8+16..+23], kh = lane>>4.
static __device__ __forceinline__ v16bf load_a_frag(const __bf16* p)
{
    v8bf lo = *(const v8bf*)p;
    v8bf hi = *(const v8bf*)(p + 16);
    return __builtin_shufflevector(lo, hi, 0,1,2,3,4,5,6,7,8,9,10,11,12,13,14,15);
}
// B 32x16 bf16 staged n-major: elems 0..7 = K[kh*8..+7], elems 8..15 = K[kh*8+16..+23] for col lane&15.
static __device__ __forceinline__ v16bf load_b_frag(const __bf16* q)
{
    v8bf lo = *(const v8bf*)q;
    v8bf hi = *(const v8bf*)(q + 16);
    return __builtin_shufflevector(lo, hi, 0,1,2,3,4,5,6,7,8,9,10,11,12,13,14,15);
}

// -------------------------- GEMM: C = A * Bt^T + bias (+relu), bf16 out ----
// 256 threads = 8 wave32; block tile 128(M) x 64(N); K-step 32;
// double-buffered LDS (one barrier per K-step), async global->LDS staging.
template <bool A_IS_F32_COLMAJOR>
__global__ __launch_bounds__(256) void gemm_bf16_kernel(
    const void* __restrict__ Aptr, long ldA, long aBatch,
    const __bf16* __restrict__ Bt, int ldB,
    const float* __restrict__ bias, int doRelu,
    __bf16* __restrict__ Cout, long ldC, long cBatch,
    int M, int Ksteps)
{
    // buf(sel) = [As|Bs] at smem + sel*BUFE; epilogue reuses smem as C tile.
    __shared__ __align__(16) __bf16 smem[2 * BUFE];

    const int  tid  = threadIdx.x;
    const long gm   = (long)blockIdx.x * 128;
    const int  gn   = blockIdx.y * 64;

    const __bf16* Arow = (const __bf16*)Aptr + (size_t)blockIdx.z * aBatch;
    const float*  Acol = (const float*)Aptr + (size_t)blockIdx.z * aBatch;
    __bf16* Cb = Cout + (size_t)blockIdx.z * cBatch;

    const int wave = tid >> 5, lane = tid & 31;
    const int wm = (wave & 3) * 32;          // 4 waves along M
    const int wn = (wave >> 2) * 32;         // 2 waves along N
    const int rowl = lane & 15, kh = lane >> 4;

    v8f acc[2][2] = {};

    // prefetch registers (manual paths)
    float4 pf_colA0, pf_colA1, pf_colA2, pf_colA3;
#if !defined(HAVE_ASYNC_LDS)
    uint4 pf_rowA0 = {}, pf_rowA1 = {}, pf_B = {};
#endif

    // runtime-offset buffer selection (no LDS pointer aggregates!)
    auto As_of = [&](int sel) -> __bf16* { return smem + sel * BUFE; };
    auto Bs_of = [&](int sel) -> __bf16* { return smem + sel * BUFE + 128 * LDSA; };

    auto stage_issue = [&](int k0, __bf16* As, __bf16* Bs) {
        // ---- A tile 128x32 ----
        if (A_IS_F32_COLMAJOR) {
            const int k = tid >> 3, part = tid & 7;
            const float* src = Acol + (long)(k0 + k) * ldA + gm + part * 16;
            pf_colA0 = ((const float4*)src)[0];
            pf_colA1 = ((const float4*)src)[1];
            pf_colA2 = ((const float4*)src)[2];
            pf_colA3 = ((const float4*)src)[3];
        } else {
#if defined(HAVE_ASYNC_LDS)
            #pragma unroll
            for (int q = 0; q < 2; ++q) {
                const int id = q * 256 + tid;
                const int r = id >> 2, seg = id & 3;
                long ar = gm + r; if (ar >= M) ar = M - 1;   // clamp; rows unused
                async_copy16(Arow + ar * ldA + k0 + seg * 8,
                             &As[r * LDSA + seg * 8]);
            }
#else
            const int r = tid >> 1, h = tid & 1;
            long ar = gm + r; if (ar >= M) ar = M - 1;
            const __bf16* src = Arow + ar * ldA + k0 + h * 16;
            pf_rowA0 = ((const uint4*)src)[0];
            pf_rowA1 = ((const uint4*)src)[1];
#endif
        }
        // ---- Bt tile 64(n)x32(k) ----
        {
            const int n = tid >> 2, seg = tid & 3;
            const __bf16* src = Bt + (long)(gn + n) * ldB + k0 + seg * 8;
#if defined(HAVE_ASYNC_LDS)
            async_copy16(src, &Bs[n * LDSB + seg * 8]);
#else
            pf_B = *(const uint4*)src;
#endif
        }
    };

    auto stage_commit = [&](__bf16* As, __bf16* Bs) {
        if (A_IS_F32_COLMAJOR) {
            const int k = tid >> 3, part = tid & 7;
            float v[16] = { pf_colA0.x, pf_colA0.y, pf_colA0.z, pf_colA0.w,
                            pf_colA1.x, pf_colA1.y, pf_colA1.z, pf_colA1.w,
                            pf_colA2.x, pf_colA2.y, pf_colA2.z, pf_colA2.w,
                            pf_colA3.x, pf_colA3.y, pf_colA3.z, pf_colA3.w };
            #pragma unroll
            for (int j = 0; j < 16; ++j)
                As[(part * 16 + j) * LDSA + k] = f2b(v[j]);
        }
#if !defined(HAVE_ASYNC_LDS)
        if (!A_IS_F32_COLMAJOR) {
            const int r = tid >> 1, h = tid & 1;
            *(uint4*)&As[r * LDSA + h * 16]     = pf_rowA0;
            *(uint4*)&As[r * LDSA + h * 16 + 8] = pf_rowA1;
        }
        {
            const int n = tid >> 2, seg = tid & 3;
            *(uint4*)&Bs[n * LDSB + seg * 8] = pf_B;
        }
#else
        (void)As; (void)Bs;
#endif
    };

    // prologue: stage tile 0 into buffer 0
    stage_issue(0, As_of(0), Bs_of(0));
    stage_commit(As_of(0), Bs_of(0));

    for (int ks = 0; ks < Ksteps; ++ks) {
        const int cur = ks & 1, nxt = cur ^ 1;
#if defined(HAVE_ASYNC_LDS)
        wait_async0();                 // our async writes to buf[cur] done
#endif
        __syncthreads();               // buf[cur] visible to all waves

        const bool more = (ks + 1 < Ksteps);
        if (more) stage_issue((ks + 1) * 32, As_of(nxt), Bs_of(nxt));

        const __bf16* As = As_of(cur);
        const __bf16* Bs = Bs_of(cur);
        v16bf a0 = load_a_frag(&As[(wm +      rowl) * LDSA + kh * 8]);
        v16bf a1 = load_a_frag(&As[(wm + 16 + rowl) * LDSA + kh * 8]);
        v16bf b0 = load_b_frag(&Bs[(wn +      rowl) * LDSB + kh * 8]);
        v16bf b1 = load_b_frag(&Bs[(wn + 16 + rowl) * LDSB + kh * 8]);

        acc[0][0] = __builtin_amdgcn_wmma_f32_16x16x32_bf16(false, a0, false, b0, (short)0, acc[0][0], false, false);
        acc[0][1] = __builtin_amdgcn_wmma_f32_16x16x32_bf16(false, a0, false, b1, (short)0, acc[0][1], false, false);
        acc[1][0] = __builtin_amdgcn_wmma_f32_16x16x32_bf16(false, a1, false, b0, (short)0, acc[1][0], false, false);
        acc[1][1] = __builtin_amdgcn_wmma_f32_16x16x32_bf16(false, a1, false, b1, (short)0, acc[1][1], false, false);

        if (more) stage_commit(As_of(nxt), Bs_of(nxt));
    }

    // ---- epilogue: bias(+relu) -> LDS transpose -> coalesced b128 stores ---
    __syncthreads();                   // done reading As/Bs everywhere
    __bf16* const Cl = smem;           // 128*64 elems <= smem size
    #pragma unroll
    for (int i = 0; i < 2; ++i) {
        #pragma unroll
        for (int j = 0; j < 2; ++j) {
            const int nl = wn + j * 16 + rowl;
            const float bv = bias[gn + nl];
            #pragma unroll
            for (int v = 0; v < 8; ++v) {
                float val = acc[i][j][v] + bv;
                if (doRelu) val = fmaxf(val, 0.0f);
                Cl[(wm + i * 16 + kh * 8 + v) * 64 + nl] = f2b(val);
            }
        }
    }
    __syncthreads();
    #pragma unroll
    for (int q = 0; q < 4; ++q) {
        const int idx = q * 256 + tid;
        const int r = idx >> 3, ch = idx & 7;
        const long mm = gm + r;
        if (mm < M)
            *(uint4*)&Cb[mm * ldC + gn + ch * 8] = *(const uint4*)&Cl[r * 64 + ch * 8];
    }
}

// -------------------------- line sampling + maxpool + concat ---------------
// x is [B][H=ix][W=iy][C=128] bf16 (channel-last, L2-resident). Each thread
// handles a channel PAIR -> b32 gathers. Block = 128 threads = 2 lines.
__global__ __launch_bounds__(128) void sample_pool_kernel(
    const __bf16* __restrict__ x, const float* __restrict__ p,
    const float* __restrict__ feat, __bf16* __restrict__ z, int L, int Mtot)
{
    const int row = blockIdx.x * 2 + (threadIdx.x >> 6);   // b*L + l
    if (row >= Mtot) return;
    const int c2 = threadIdx.x & 63;                       // channels 2c2, 2c2+1
    const int b  = row / L;

    const float* pl = p + (size_t)row * 4;
    const float ax = pl[0], ay = pl[1];   // endpoint 0
    const float bx = pl[2], by = pl[3];   // endpoint 1
    const __bf16* xb = x + ((size_t)b * HW * HW) * DIM_LOI + 2 * c2;

    float mx0[N_PTS1], mx1[N_PTS1];
    #pragma unroll
    for (int g = 0; g < N_PTS1; ++g) { mx0[g] = -3.402823466e38f; mx1[g] = -3.402823466e38f; }

    #pragma unroll 4
    for (int i = 0; i < N_PTS0; ++i) {
        const float lam = (float)i * (1.0f / (N_PTS0 - 1));
        const float px = ax * lam + bx * (1.0f - lam) - 0.5f;
        const float py = ay * lam + by * (1.0f - lam) - 0.5f;
        const float fx0 = fminf(fmaxf(floorf(px), 0.0f), (float)(HW - 1));
        const float fy0 = fminf(fmaxf(floorf(py), 0.0f), (float)(HW - 1));
        const float fx1 = fminf(fx0 + 1.0f, (float)(HW - 1));
        const float fy1 = fminf(fy0 + 1.0f, (float)(HW - 1));
        const int ix0 = (int)fx0, iy0 = (int)fy0, ix1 = (int)fx1, iy1 = (int)fy1;
        const float w00 = (fx1 - px) * (fy1 - py);
        const float w10 = (px - fx0) * (fy1 - py);
        const float w01 = (fx1 - px) * (py - fy0);
        const float w11 = (px - fx0) * (py - fy0);
        union Pk { unsigned u; __bf16 h[2]; } q00, q10, q01, q11;
        q00.u = *(const unsigned*)(xb + ((ix0 * HW + iy0) << 7));
        q10.u = *(const unsigned*)(xb + ((ix1 * HW + iy0) << 7));
        q01.u = *(const unsigned*)(xb + ((ix0 * HW + iy1) << 7));
        q11.u = *(const unsigned*)(xb + ((ix1 * HW + iy1) << 7));
        const float v0 = w00 * b2f(q00.h[0]) + w10 * b2f(q10.h[0]) +
                         w01 * b2f(q01.h[0]) + w11 * b2f(q11.h[0]);
        const float v1 = w00 * b2f(q00.h[1]) + w10 * b2f(q10.h[1]) +
                         w01 * b2f(q01.h[1]) + w11 * b2f(q11.h[1]);
        mx0[i >> 2] = fmaxf(mx0[i >> 2], v0);
        mx1[i >> 2] = fmaxf(mx1[i >> 2], v1);
    }

    // z row: [0..1023]=pooled (c*8+g), [1024..1031]=feat, [1032..1055]=0
    union { uint4 v[2]; __bf16 h[16]; } ou;
    #pragma unroll
    for (int g = 0; g < N_PTS1; ++g) { ou.h[g] = f2b(mx0[g]); ou.h[8 + g] = f2b(mx1[g]); }
    __bf16* zr = z + (size_t)row * KZ + c2 * 16;
    ((uint4*)zr)[0] = ou.v[0];
    ((uint4*)zr)[1] = ou.v[1];
    __bf16* zt = z + (size_t)row * KZ;
    if (c2 < 8)       zt[1024 + c2] = f2b(feat[(size_t)row * 8 + c2]);
    else if (c2 < 32) zt[1024 + c2] = (__bf16)0.0f;
}

// -------------------------- final 1024 -> 1 (wave-per-row dot) -------------
__global__ __launch_bounds__(256) void final_fc_kernel(
    const __bf16* __restrict__ h2, const float* __restrict__ W3,
    const float* __restrict__ b3, float* __restrict__ out, int M)
{
    const int wave = threadIdx.x >> 5, lane = threadIdx.x & 31;
    const int row = blockIdx.x * 8 + wave;
    if (row >= M) return;
    const __bf16* hr = h2 + (size_t)row * DIM_FC;
    float s = 0.0f;
    #pragma unroll 8
    for (int k = lane; k < DIM_FC; k += 32)
        s += b2f(hr[k]) * W3[k];
    #pragma unroll
    for (int off = 16; off > 0; off >>= 1)
        s += __shfl_xor(s, off, 32);
    if (lane == 0) out[row] = s + b3[0];
}

// ---------------------------------------------------------------------------
static inline size_t align256(size_t x) { return (x + 255) & ~(size_t)255; }

extern "C" void kernel_launch(void* const* d_in, const int* in_sizes, int n_in,
                              void* d_out, int out_size, void* d_ws, size_t ws_size,
                              hipStream_t stream) {
    (void)n_in; (void)out_size; (void)ws_size;
    const float* feature = (const float*)d_in[0];   // [B,256,128,128]
    const float* p       = (const float*)d_in[1];   // [B,L,2,2]
    const float* feat    = (const float*)d_in[2];   // [B,L,8]
    const float* W_fc1   = (const float*)d_in[3];   // [128,256]
    const float* b_fc1   = (const float*)d_in[4];   // [128]
    const float* W1      = (const float*)d_in[5];   // [1032,1024]
    const float* b1      = (const float*)d_in[6];   // [1024]
    const float* W2      = (const float*)d_in[7];   // [1024,1024]
    const float* b2      = (const float*)d_in[8];   // [1024]
    const float* W3      = (const float*)d_in[9];   // [1024,1]
    const float* b3      = (const float*)d_in[10];  // [1]
    float* logits = (float*)d_out;

    const int B = in_sizes[0] / (256 * HW * HW);
    const int L = in_sizes[1] / (B * 4);
    const int M = B * L;                            // 40000

    // workspace carve-up (bf16 buffers)
    char* ws = (char*)d_ws;
    size_t off = 0;
    __bf16* wfc1b = (__bf16*)(ws + off); off = align256(off + (size_t)DIM_LOI * 256 * 2);
    __bf16* w1t   = (__bf16*)(ws + off); off = align256(off + (size_t)DIM_FC * KZ * 2);
    __bf16* w2t   = (__bf16*)(ws + off); off = align256(off + (size_t)DIM_FC * DIM_FC * 2);
    __bf16* xbuf  = (__bf16*)(ws + off); off = align256(off + (size_t)B * HW * HW * DIM_LOI * 2);
    __bf16* zbuf  = (__bf16*)(ws + off); off = align256(off + (size_t)M * KZ * 2);
    __bf16* h1    = (__bf16*)(ws + off); off = align256(off + (size_t)M * DIM_FC * 2);
    __bf16* h2    = (__bf16*)(ws + off); off = align256(off + (size_t)M * DIM_FC * 2);

    // 1) weights -> bf16 (transposed for B-fragment contiguity)
    {
        const int total = DIM_LOI * 256 + DIM_FC * KZ + DIM_FC * DIM_FC;
        prep_weights_kernel<<<(total + 255) / 256, 256, 0, stream>>>(
            W_fc1, W1, W2, wfc1b, w1t, w2t);
    }

    // 2) fc1: x[b][p][d] = sum_c feature[b][c][p] * W_fc1[d][c] + b_fc1[d]
    {
        dim3 grid(HW * HW / 128, DIM_LOI / 64, B);
        gemm_bf16_kernel<true><<<grid, 256, 0, stream>>>(
            (const void*)feature, (long)(HW * HW), (long)256 * HW * HW,
            wfc1b, 256, b_fc1, /*relu=*/0,
            xbuf, (long)DIM_LOI, (long)HW * HW * DIM_LOI,
            HW * HW, 256 / 32);
    }

    // 3) bilinear line sampling + maxpool4 + concat feat -> z [M][1056]
    sample_pool_kernel<<<(M + 1) / 2, 128, 0, stream>>>(xbuf, p, feat, zbuf, L, M);

    // 4) h1 = relu(z @ W1 + b1)
    {
        dim3 grid((M + 127) / 128, DIM_FC / 64, 1);
        gemm_bf16_kernel<false><<<grid, 256, 0, stream>>>(
            (const void*)zbuf, (long)KZ, 0L,
            w1t, KZ, b1, /*relu=*/1,
            h1, (long)DIM_FC, 0L, M, KZ / 32);
    }

    // 5) h2 = relu(h1 @ W2 + b2)
    {
        dim3 grid((M + 127) / 128, DIM_FC / 64, 1);
        gemm_bf16_kernel<false><<<grid, 256, 0, stream>>>(
            (const void*)h1, (long)DIM_FC, 0L,
            w2t, DIM_FC, b2, /*relu=*/1,
            h2, (long)DIM_FC, 0L, M, DIM_FC / 32);
    }

    // 6) logits = h2 @ W3 + b3
    final_fc_kernel<<<(M + 7) / 8, 256, 0, stream>>>(h2, W3, b3, logits, M);
}
